// SparseLinear_65609920413833
// MI455X (gfx1250) — matrix operand order repaired and checked by reading, not compiled, not run on hardware
//
#include <hip/hip_runtime.h>

// ---------------------------------------------------------------------------
// SparseLinear forward on gfx1250: out = x @ w^T + bias
//   x: [N, C] f32, w: [K, C] f32 (block sparse), bias: [K] f32, out: [N,K] f32
//
// Roofline (MI455X, 23.3 TB/s): mandatory traffic = x(128MB f32) + out(128MB)
// + w(4MB, L2-resident) ~= 260 MB -> ~11 us floor. bf16 WMMA compute lands in
// the same ballpark, f32 WMMA (K=4/instr) would be 8x more matrix issue ->
// dense bf16 WMMA with on-the-fly f32->bf16 (v_cvt_pk_bf16_f32), double-
// buffered LDS, 1 barrier per k-step. The 1x16 block sparsity (80%) cannot
// map to SWMMAC 2:4 and the kernel is bound on x/out traffic, so dense wins.
//
// Bias is folded into the accumulator init (C/D layout: all 8 acc elements of
// a tile share the same output column), so the epilogue is pure stores.
// ---------------------------------------------------------------------------

typedef __bf16 bf16;
typedef __attribute__((ext_vector_type(4)))  float  f32x4;
typedef __attribute__((ext_vector_type(4)))  __bf16 bf16x4;
typedef __attribute__((ext_vector_type(8)))  __bf16 bf16x8;
typedef __attribute__((ext_vector_type(16))) __bf16 v16bf;
typedef __attribute__((ext_vector_type(8)))  float  v8f;

#define TILE_M 128   // rows of x per workgroup
#define TILE_N 128   // output features per workgroup
#define TILE_K 32    // contraction step (= WMMA bf16 K)
#define LDT    40    // padded LDS row stride in bf16 elems (80 B, 16B-aligned)

__global__ __launch_bounds__(256)
void sparse_linear_wmma_bf16(const float* __restrict__ X,
                             const float* __restrict__ W,
                             const float* __restrict__ BIAS,
                             float* __restrict__ OUT,
                             int N, int C, int K) {
  __shared__ bf16 As[2][TILE_M * LDT];   // x tile  [m][c]  (bf16)
  __shared__ bf16 Bs[2][TILE_N * LDT];   // w tile  [k][c]  (bf16)

  const int tid  = threadIdx.x;
  const int lane = tid & 31;
  const int wv   = tid >> 5;    // wave 0..7
  const int wm   = wv & 3;      // 4 wave-rows  -> 32 output rows each
  const int wn   = wv >> 2;     // 2 wave-cols  -> 64 output cols each

  const int rowBase = blockIdx.x * TILE_M;
  const int colBase = blockIdx.y * TILE_N;

  // ---- accumulators, initialized with broadcast bias --------------------
  // C/D layout: lane L -> column n = L%16; VGPR r -> row m = r + (L>=16)*8.
  // bias depends only on the column, so all 8 elements get the same value.
  const int nl = lane & 15;
  v8f acc[2][4];
#pragma unroll
  for (int nt = 0; nt < 4; ++nt) {
    const float bv = BIAS[colBase + wn * 64 + nt * 16 + nl];
    const v8f bini = (v8f){bv, bv, bv, bv, bv, bv, bv, bv};
    acc[0][nt] = bini;
    acc[1][nt] = bini;
  }

  // ---- global tile staging (4 float4 of A + 4 float4 of B per thread) ----
  f32x4 ar[4], br[4];
  const int T = C / TILE_K;   // 32 k-steps

  auto loadTiles = [&](int ct) {
    const int cb = ct * TILE_K;
#pragma unroll
    for (int i = 0; i < 4; ++i) {
      int q    = i * 256 + tid;      // 0..1023
      int row  = q >> 3;             // 0..127 (tile row)
      int col4 = q & 7;              // 8 float4 per 32-float row
      ar[i] = *(const f32x4*)(X + (size_t)(rowBase + row) * C + cb + col4 * 4);
      br[i] = *(const f32x4*)(W + (size_t)(colBase + row) * C + cb + col4 * 4);
    }
    if (ct + 2 < T) {  // speculative L2 prefetch two steps ahead
      const int pb = (ct + 2) * TILE_K;
      __builtin_prefetch(X + (size_t)(rowBase + (tid >> 1)) * C + pb, 0, 0);
      __builtin_prefetch(W + (size_t)(colBase + (tid >> 1)) * C + pb, 0, 0);
    }
  };

  auto stash = [&](int buf) {
#pragma unroll
    for (int i = 0; i < 4; ++i) {
      int q    = i * 256 + tid;
      int row  = q >> 3;
      int col4 = q & 7;
      bf16x4 a, b;
#pragma unroll
      for (int e = 0; e < 4; ++e) {
        // lowers to v_cvt_pk_bf16_f32 (verified in round-2 asm)
        a[e] = (bf16)ar[i][e];
        b[e] = (bf16)br[i][e];
      }
      *(bf16x4*)&As[buf][row * LDT + col4 * 4] = a;
      *(bf16x4*)&Bs[buf][row * LDT + col4 * 4] = b;
    }
  };

  // ---- per-lane fragment addressing (CDNA5 16-bit WMMA layouts) ----
  // A 16x32: lanes 0-15 -> M=lane, elems 0..7 = K 0..7,  8..15 = K 16..23
  //          lanes16-31 -> M=lane-16, elems 0..7 = K 8..15, 8..15 = K 24..31
  const int mrow0 = wm * 32 + (lane & 15);
  const int c0a   = (lane >> 4) * 8;
  // B 32x16: same K order as A: lanes 0-15 -> elems 0..7 = K 0..7, 8..15 = K 16..23;
  //          lanes 16-31 -> elems 0..7 = K 8..15, 8..15 = K 24..31
  const int kcol0 = wn * 64 + (lane & 15);
  const int c0b   = (lane >> 4) * 8;

  auto compute = [&](int buf) {
    v16bf af[2], bfr[4];
#pragma unroll
    for (int mt = 0; mt < 2; ++mt) {
      const bf16* p = &As[buf][(mrow0 + mt * 16) * LDT + c0a];
      bf16x8 lo = *(const bf16x8*)p;
      bf16x8 hi = *(const bf16x8*)(p + 16);
      af[mt] = __builtin_shufflevector(lo, hi, 0, 1, 2, 3, 4, 5, 6, 7,
                                               8, 9, 10, 11, 12, 13, 14, 15);
    }
#pragma unroll
    for (int nt = 0; nt < 4; ++nt) {
      const bf16* p = &Bs[buf][(kcol0 + nt * 16) * LDT + c0b];
      bf16x8 lo = *(const bf16x8*)p;
      bf16x8 hi = *(const bf16x8*)(p + 16);
      bfr[nt] = __builtin_shufflevector(lo, hi, 0, 1, 2, 3, 4, 5, 6, 7,
                                                8, 9, 10, 11, 12, 13, 14, 15);
    }
#pragma unroll
    for (int mt = 0; mt < 2; ++mt)
#pragma unroll
      for (int nt = 0; nt < 4; ++nt)
        acc[mt][nt] = __builtin_amdgcn_wmma_f32_16x16x32_bf16(
            /*neg_a=*/false, af[mt], /*neg_b=*/false, bfr[nt],
            /*c_mod=*/(short)0, acc[mt][nt],
            /*reuse_a=*/false, /*reuse_b=*/false);
  };

  // ---- software-pipelined main loop: 1 barrier / k-step, double buffer ----
  loadTiles(0);
  for (int t = 0; t < T; ++t) {
    const int buf = t & 1;
    stash(buf);
    __syncthreads();
    if (t + 1 < T) loadTiles(t + 1);   // hide HBM latency behind WMMAs
    compute(buf);
  }

  // ---- epilogue: pure stores (bias already inside the accumulators) ----
  const int mo = (lane >> 4) * 8;
#pragma unroll
  for (int mt = 0; mt < 2; ++mt)
#pragma unroll
    for (int nt = 0; nt < 4; ++nt) {
      const int kg = colBase + wn * 64 + nt * 16 + nl;
#pragma unroll
      for (int r = 0; r < 8; ++r) {
        const int rg = rowBase + wm * 32 + mt * 16 + mo + r;
        OUT[(size_t)rg * K + kg] = acc[mt][nt][r];
      }
    }
}

extern "C" void kernel_launch(void* const* d_in, const int* in_sizes, int n_in,
                              void* d_out, int out_size, void* d_ws, size_t ws_size,
                              hipStream_t stream) {
  const float* x    = (const float*)d_in[0];
  const float* w    = (const float*)d_in[1];
  const float* bias = (const float*)d_in[2];
  float* out        = (float*)d_out;

  const int K = in_sizes[2];             // 1024
  const int C = in_sizes[1] / K;         // 1024
  const int N = in_sizes[0] / C;         // 32768

  dim3 grid(N / TILE_M, K / TILE_N);     // (256, 8)
  sparse_linear_wmma_bf16<<<grid, dim3(256), 0, stream>>>(x, w, bias, out, N, C, K);
}
